// PSNet_1331439861912
// MI455X (gfx1250) — matrix operand order repaired and checked by reading, not compiled, not run on hardware
//
#include <hip/hip_runtime.h>

typedef __attribute__((ext_vector_type(16))) _Float16 v16h;
typedef __attribute__((ext_vector_type(8)))  _Float16 v8h;
typedef __attribute__((ext_vector_type(8)))  float    v8f;
typedef __attribute__((ext_vector_type(8)))  unsigned v8u;

#define BN_INV_C 0.9999950000374997f

union AU { v16h v16; v8h v8[2]; };
union BU { v16h v16; v8u u8; };
union HP { _Float16 h[2]; unsigned u; };

struct ConvP {
  const _Float16* __restrict__ in;   // (D,H,W,Cpi) channel-last f16, Cpi%32==0
  const unsigned* __restrict__ wp;   // packed B fragments [ks][ntile][lane][8]
  const float* __restrict__ scale;   // BN gamma (or null)
  const float* __restrict__ bias;    // BN beta
  const _Float16* __restrict__ res;  // residual (Do,Ho,Wo,Cpo) (or null)
  _Float16* __restrict__ out16;      // (Do,Ho,Wo,Cpo) (or null)
  float* __restrict__ out32;         // accumulate (+=) channel 0 only (or null)
  int D, H, W, Cpi;
  int Do, Ho, Wo;
  int Cout, Cpo;
  int KD, KH, KW;
  int dil, stride, act;  // act: 0 none, 1 relu, 2 leaky(0.1)
  float scaleMul;        // BN_INV or 1
};

// ---------------------------------------------------------------------------
// Implicit-GEMM conv using v_wmma_f32_16x16x32_f16, K ordered (td,th,tw,ci).
// A fragment (16x32 f16): lanes 0-15: M=lane, ci c0+0..15;
//   lanes 16-31: M=lane-16, ci c0+16..31
//   -> two contiguous 16B global_load_b128 per lane per K-step.
// B fragment: pre-swizzled per-lane in memory -> one 32B global_load_b256.
// C/D fragment: VGPR r, lanes 0-15: M=m0+r, N=lane; lanes 16-31: M=m0+8+r.
// ---------------------------------------------------------------------------
template <int NT>
__global__ void __launch_bounds__(256) conv_wmma(ConvP p) {
  const int lane = threadIdx.x & 31;
  const int wv   = threadIdx.x >> 5;
  const int Mtot = p.Do * p.Ho * p.Wo;
  const int mtiles = (Mtot + 15) >> 4;
  const int mtile = blockIdx.x * 8 + wv;
  if (mtile >= mtiles) return;
  const int m0 = mtile << 4;
  const int nlane = lane & 15;

  // Per-lane A row (output position). Both lane halves cover M = m0..m0+15.
  const int lm = m0 + nlane;
  const bool mvalid = lm < Mtot;
  const int mm = mvalid ? lm : 0;
  const int ow = mm % p.Wo;
  const int t1 = mm / p.Wo;
  const int oh = t1 % p.Ho;
  const int od = t1 / p.Ho;

  const int cchunks = p.Cpi >> 5;
  const int coff = (lane < 16) ? 0 : 16;
  const int ph = ((p.KH - 1) * p.dil) >> 1;
  const int pw = ((p.KW - 1) * p.dil) >> 1;
  const int pd = (p.KD - 1) >> 1;
  const int ohs = oh * p.stride - ph;
  const int ows = ow * p.stride - pw;

  const v16h zero16 = {};
  v8f acc[NT];
#pragma unroll
  for (int j = 0; j < NT; ++j) acc[j] = (v8f){};

  // B fragment pointer: lane's 8 dwords are contiguous; advance per K-step.
  const unsigned* __restrict__ wp0 = p.wp;
  size_t kofs = (size_t)lane * 8;

  for (int td = 0; td < p.KD; ++td) {
    const int id = od + td - pd;
    for (int th = 0; th < p.KH; ++th) {
      const int ih = ohs + th * p.dil;
      for (int tw = 0; tw < p.KW; ++tw) {
        const int iw = ows + tw * p.dil;
        const bool ok = mvalid && (unsigned)iw < (unsigned)p.W &&
                        (unsigned)ih < (unsigned)p.H &&
                        (unsigned)id < (unsigned)p.D;
        const _Float16* __restrict__ abase =
            p.in + (size_t)((id * p.H + ih) * p.W + iw) * p.Cpi + coff;
        for (int cc = 0; cc < cchunks; ++cc) {
          AU a;
          if (ok) {
            a.v8[0] = *(const v8h*)(abase);       // ci c0..c0+7
            a.v8[1] = *(const v8h*)(abase + 8);  // ci c0+8..c0+15
          } else {
            a.v16 = zero16;
          }
          abase += 32;
          __builtin_prefetch(wp0 + kofs + (size_t)NT * 256, 0, 1);
#pragma unroll
          for (int j = 0; j < NT; ++j) {
            BU b;
            b.u8 = *(const v8u*)(wp0 + kofs + (size_t)j * 256);
            acc[j] = __builtin_amdgcn_wmma_f32_16x16x32_f16(
                false, a.v16, false, b.v16, (short)0, acc[j], false, false);
          }
          kofs += (size_t)NT * 256;
        }
      }
    }
  }

  // Epilogue
  const int mb = m0 + ((lane < 16) ? 0 : 8);
#pragma unroll
  for (int j = 0; j < NT; ++j) {
    const int n = (j << 4) + nlane;
    const bool nok = n < p.Cout;
    float sc = 1.f, bs = 0.f;
    if (p.scale && nok) { sc = p.scale[n] * p.scaleMul; bs = p.bias[n]; }
#pragma unroll
    for (int r = 0; r < 8; ++r) {
      const int m = mb + r;
      if (m >= Mtot) continue;
      float y = acc[j][r];
      if (p.scale) y = y * sc + bs;
      if (p.act == 1) y = fmaxf(y, 0.f);
      else if (p.act == 2) y = (y > 0.f) ? y : 0.1f * y;
      if (p.out32) {
        if (n == 0) p.out32[m] += y;
      } else if (nok) {
        if (p.res) y += (float)p.res[(size_t)m * p.Cpo + n];
        p.out16[(size_t)m * p.Cpo + n] = (_Float16)y;
      }
    }
  }
}

// --------------------------- prep / glue kernels ---------------------------

// OIHW/OIDHW f32 weights -> B fragments in per-lane order:
// wp[((ks*ntiles + j)*32 + lane)*8 + v] = dword {k=2*k2, k=2*k2+1} at col n,
// where k2 = ks*16 + (lane<16 ? 0 : 8) + v, n = j*16 + (lane&15),
// K ordered (tap, ci) with ci padded to Cpi.
__global__ void pack_weights_k(const float* __restrict__ w,
                               unsigned* __restrict__ wp, int Cin, int Cpi,
                               int Cout, int Np, int kvol, int tot) {
  int idx = blockIdx.x * blockDim.x + threadIdx.x;
  if (idx >= tot) return;
  int v = idx & 7;
  int lane = (idx >> 3) & 31;
  int rest = idx >> 8;
  int ntiles = Np >> 4;
  int j = rest % ntiles;
  int ks = rest / ntiles;
  int n = (j << 4) + (lane & 15);
  int k2 = (ks << 4) + ((lane < 16) ? 0 : 8) + v;
  HP r;
#pragma unroll
  for (int i = 0; i < 2; ++i) {
    int k = 2 * k2 + i;
    int tap = k / Cpi;
    int ci = k - tap * Cpi;
    float val = 0.f;
    if (ci < Cin && n < Cout) val = w[((size_t)n * Cin + ci) * kvol + tap];
    r.h[i] = (_Float16)val;
  }
  wp[idx] = r.u;
}

// f32 CHW -> f16 HWC (channel padded to Cp with zeros)
__global__ void chlast_k(const float* __restrict__ src,
                         _Float16* __restrict__ dst, int HW, int C, int Cp) {
  int idx = blockIdx.x * blockDim.x + threadIdx.x;
  if (idx >= HW * Cp) return;
  int c = idx % Cp;
  int pix = idx / Cp;
  float v = (c < C) ? src[(size_t)c * HW + pix] : 0.f;
  dst[idx] = (_Float16)v;
}

// Build cost volume slice: channels 0-31 = ref feature, 32-63 = warped target.
__global__ void warp_cost_k(const _Float16* __restrict__ reff,
                            const _Float16* __restrict__ tfea,
                            const float* __restrict__ pose,
                            const float* __restrict__ Km,
                            const float* __restrict__ Kim,
                            _Float16* __restrict__ cost, int tgt, int L, int h,
                            int w) {
  int idx = blockIdx.x * blockDim.x + threadIdx.x;
  if (idx >= L * h * w) return;
  int x = idx % w;
  int t = idx / w;
  int y = t % h;
  int l = t / h;

  float K4[9], Ki[9];
#pragma unroll
  for (int i = 0; i < 9; ++i) { K4[i] = Km[i]; Ki[i] = Kim[i]; }
  K4[0] *= 0.25f; K4[1] *= 0.25f; K4[2] *= 0.25f;
  K4[3] *= 0.25f; K4[4] *= 0.25f; K4[5] *= 0.25f;
  Ki[0] *= 4.f; Ki[1] *= 4.f; Ki[3] *= 4.f; Ki[4] *= 4.f;
  const float* P = pose + tgt * 12;

  float depth = 32.0f / ((float)l + 1e-16f);
  float px = (float)x, py = (float)y;
  float cx = (Ki[0] * px + Ki[1] * py + Ki[2]) * depth;
  float cy = (Ki[3] * px + Ki[4] * py + Ki[5]) * depth;
  float cz = (Ki[6] * px + Ki[7] * py + Ki[8]) * depth;
  float X = P[0] * cx + P[1] * cy + P[2] * cz + P[3];
  float Y = P[4] * cx + P[5] * cy + P[6] * cz + P[7];
  float Z = P[8] * cx + P[9] * cy + P[10] * cz + P[11];
  float pu = K4[0] * X + K4[1] * Y + K4[2] * Z;
  float pv = K4[3] * X + K4[4] * Y + K4[5] * Z;
  float pz = fmaxf(K4[6] * X + K4[7] * Y + K4[8] * Z, 0.001f);
  float u = pu / pz, v = pv / pz;

  float x0f = floorf(u), y0f = floorf(v);
  float fx = u - x0f, fy = v - y0f;
  const float dx[4] = {0.f, 1.f, 0.f, 1.f};
  const float dy[4] = {0.f, 0.f, 1.f, 1.f};
  const float bw[4] = {(1.f - fx) * (1.f - fy), fx * (1.f - fy),
                       (1.f - fx) * fy, fx * fy};
  float cw[4];
  int co[4];
#pragma unroll
  for (int k = 0; k < 4; ++k) {
    float xx = x0f + dx[k], yy = y0f + dy[k];
    float m = (xx >= 0.f && xx <= (float)(w - 1) && yy >= 0.f &&
               yy <= (float)(h - 1)) ? 1.f : 0.f;
    float xc = fminf(fmaxf(xx, 0.f), (float)(w - 1));
    float yc = fminf(fmaxf(yy, 0.f), (float)(h - 1));
    co[k] = ((int)yc * w + (int)xc) * 32;
    cw[k] = bw[k] * m;
  }

  _Float16* dst = cost + (size_t)idx * 64;
  const _Float16* rf = reff + (size_t)(y * w + x) * 32;
#pragma unroll 8
  for (int c = 0; c < 32; ++c) dst[c] = rf[c];
#pragma unroll 4
  for (int c = 0; c < 32; ++c) {
    float vv = cw[0] * (float)tfea[co[0] + c] + cw[1] * (float)tfea[co[1] + c] +
               cw[2] * (float)tfea[co[2] + c] + cw[3] * (float)tfea[co[3] + c];
    dst[32 + c] = (_Float16)vv;
  }
}

// Context input: ch 0-31 ref feature, ch 32 = 0.5*costs, ch 33-63 zero.
__global__ void ctx_input_k(const _Float16* __restrict__ reff,
                            const float* __restrict__ costs,
                            _Float16* __restrict__ dst, int L, int hw) {
  int idx = blockIdx.x * blockDim.x + threadIdx.x;
  if (idx >= L * hw * 64) return;
  int c = idx % 64;
  int rest = idx / 64;
  int pix = rest % hw;
  float v = 0.f;
  if (c < 32) v = (float)reff[(size_t)pix * 32 + c];
  else if (c == 32) v = 0.5f * costs[rest];
  dst[idx] = (_Float16)v;
}

__global__ void scale_init_k(float* __restrict__ dst,
                             const float* __restrict__ src, float s, int n) {
  int i = blockIdx.x * blockDim.x + threadIdx.x;
  if (i >= n) return;
  dst[i] = src ? src[i] * s : 0.f;
}

// Bilinear x4 upsample (half-pixel) + softmax over L + disparity -> depth
__global__ void final_k(const float* __restrict__ refined,
                        float* __restrict__ out, int L, int hh, int ww, int H,
                        int W) {
  int idx = blockIdx.x * blockDim.x + threadIdx.x;
  if (idx >= H * W) return;
  int x = idx % W, y = idx / W;
  float sx = (x + 0.5f) * ((float)ww / (float)W) - 0.5f;
  float sy = (y + 0.5f) * ((float)hh / (float)H) - 0.5f;
  float x0f = floorf(sx), y0f = floorf(sy);
  float fx = sx - x0f, fy = sy - y0f;
  int x0 = (int)x0f, y0 = (int)y0f;
  int x1 = x0 + 1, y1 = y0 + 1;
  x0 = x0 < 0 ? 0 : (x0 > ww - 1 ? ww - 1 : x0);
  x1 = x1 < 0 ? 0 : (x1 > ww - 1 ? ww - 1 : x1);
  y0 = y0 < 0 ? 0 : (y0 > hh - 1 ? hh - 1 : y0);
  y1 = y1 < 0 ? 0 : (y1 > hh - 1 ? hh - 1 : y1);
  float w00 = (1.f - fx) * (1.f - fy), w10 = fx * (1.f - fy);
  float w01 = (1.f - fx) * fy, w11 = fx * fy;
  int o00 = y0 * ww + x0, o10 = y0 * ww + x1;
  int o01 = y1 * ww + x0, o11 = y1 * ww + x1;

  float vals[64];
  float mx = -3.4e38f;
  for (int l = 0; l < L; ++l) {
    const float* s = refined + (size_t)l * hh * ww;
    float v = w00 * s[o00] + w10 * s[o10] + w01 * s[o01] + w11 * s[o11];
    vals[l] = v;
    mx = fmaxf(mx, v);
  }
  float se = 0.f, sw = 0.f;
  for (int l = 0; l < L; ++l) {
    float e = __expf(vals[l] - mx);
    se += e;
    sw += e * (float)l;
  }
  float disp = sw / se;
  out[idx] = 32.0f / (disp + 1e-16f);
}

// --------------------------------- driver ----------------------------------

extern "C" void kernel_launch(void* const* d_in, const int* in_sizes, int n_in,
                              void* d_out, int out_size, void* d_ws,
                              size_t ws_size, hipStream_t stream) {
  (void)in_sizes; (void)n_in; (void)out_size; (void)ws_size;
  const int h = 64, w = 80, L = 64, Himg = 256, Wimg = 320;
  const int HWs = h * w;    // 5120
  const int VOX = L * HWs;  // 327680

  const float* ref     = (const float*)d_in[0];
  const float* targets = (const float*)d_in[1];
  const float* pose    = (const float*)d_in[2];
  const float* Kmat    = (const float*)d_in[3];
  const float* Kinv    = (const float*)d_in[4];
  // params pytree (keys sorted): cls, ctx, d0..d4, feat
  const float* cls_w1 = (const float*)d_in[5];
  const float* cls_g  = (const float*)d_in[6];
  const float* cls_b  = (const float*)d_in[7];
  const float* cls_w2 = (const float*)d_in[8];
  const float* ctx_w[7];
  for (int i = 0; i < 7; ++i) ctx_w[i] = (const float*)d_in[9 + i];
  const float* dw[5][6];  // d0..d4
  for (int g = 0; g < 5; ++g)
    for (int i = 0; i < 6; ++i) dw[g][i] = (const float*)d_in[16 + g * 6 + i];
  const float* feat_w[4];
  for (int i = 0; i < 4; ++i) feat_w[i] = (const float*)d_in[46 + i];

  // ---- workspace allocator ----
  size_t off = 0;
  auto alloc = [&](size_t bytes) -> char* {
    off = (off + 255) & ~(size_t)255;
    char* p = (char*)d_ws + off;
    off += bytes;
    return p;
  };
  auto wsz = [](int Cpi, int Np, int kvol) {
    return (size_t)(kvol * Cpi / 2) * Np * 4;
  };

  unsigned* wpF[4];
  for (int i = 0; i < 4; ++i) wpF[i] = (unsigned*)alloc(wsz(32, 32, 9));
  unsigned* wpD0a = (unsigned*)alloc(wsz(64, 32, 27));
  unsigned* wpD0b = (unsigned*)alloc(wsz(32, 32, 27));
  unsigned* wpDr[4][2];
  for (int r = 0; r < 4; ++r)
    for (int i = 0; i < 2; ++i) wpDr[r][i] = (unsigned*)alloc(wsz(32, 32, 27));
  unsigned* wpCls1 = (unsigned*)alloc(wsz(32, 32, 27));
  unsigned* wpCls2 = (unsigned*)alloc(wsz(32, 16, 27));
  const int ctxCpi[7] = {64, 128, 128, 128, 96, 64, 32};
  const int ctxNp[7]  = {128, 128, 128, 96, 64, 32, 16};
  const int ctxCout[7] = {128, 128, 128, 96, 64, 32, 1};
  const int ctxCin[7] = {33, 128, 128, 128, 96, 64, 32};
  const int ctxDil[7] = {1, 2, 4, 8, 16, 1, 1};
  unsigned* wpCtx[7];
  for (int i = 0; i < 7; ++i)
    wpCtx[i] = (unsigned*)alloc(wsz(ctxCpi[i], ctxNp[i], 9));

  _Float16* bufImg = (_Float16*)alloc((size_t)Himg * Wimg * 32 * 2);
  _Float16* pA = (_Float16*)alloc((size_t)128 * 160 * 32 * 2);
  _Float16* pB = (_Float16*)alloc((size_t)128 * 160 * 32 * 2);
  _Float16* pC = (_Float16*)alloc((size_t)HWs * 32 * 2);
  _Float16* feats[3];
  for (int i = 0; i < 3; ++i) feats[i] = (_Float16*)alloc((size_t)HWs * 32 * 2);
  _Float16* costvol = (_Float16*)alloc((size_t)VOX * 64 * 2);
  _Float16* volA = (_Float16*)alloc((size_t)VOX * 32 * 2);
  _Float16* volB = (_Float16*)alloc((size_t)VOX * 32 * 2);
  float* costs = (float*)alloc((size_t)VOX * 4);
  float* refined = (float*)alloc((size_t)VOX * 4);

  // ---- conv launcher (dispatch on N tiles) ----
  auto conv = [&](const _Float16* in, const unsigned* wp, const float* g,
                  const float* bb, const _Float16* res, _Float16* out16,
                  float* out32, int D, int Hh, int Ww, int Cpi, int Do, int Ho,
                  int Wo, int Cout, int Cpo, int Np, int KD, int KH, int KW,
                  int dil, int stride, int act, float sm) {
    ConvP p{in, wp, g, bb, res, out16, out32, D, Hh, Ww, Cpi, Do, Ho, Wo,
            Cout, Cpo, KD, KH, KW, dil, stride, act, sm};
    int Mtot = Do * Ho * Wo;
    int grid = ((Mtot + 15) / 16 + 7) / 8;
    switch (Np / 16) {
      case 1: conv_wmma<1><<<grid, 256, 0, stream>>>(p); break;
      case 2: conv_wmma<2><<<grid, 256, 0, stream>>>(p); break;
      case 4: conv_wmma<4><<<grid, 256, 0, stream>>>(p); break;
      case 6: conv_wmma<6><<<grid, 256, 0, stream>>>(p); break;
      case 8: conv_wmma<8><<<grid, 256, 0, stream>>>(p); break;
      default: break;
    }
  };
  auto pack = [&](const float* ww, unsigned* dst, int Cin, int Cpi, int Cout,
                  int Np, int kvol) {
    int tot = (kvol * Cpi / 2) * Np;
    pack_weights_k<<<(tot + 255) / 256, 256, 0, stream>>>(ww, dst, Cin, Cpi,
                                                          Cout, Np, kvol, tot);
  };

  // ---- pack all weights ----
  pack(feat_w[0], wpF[0], 3, 32, 32, 32, 9);
  for (int i = 1; i < 4; ++i) pack(feat_w[i], wpF[i], 32, 32, 32, 32, 9);
  pack(dw[0][0], wpD0a, 64, 64, 32, 32, 27);
  pack(dw[0][3], wpD0b, 32, 32, 32, 32, 27);
  for (int r = 0; r < 4; ++r) {
    pack(dw[1 + r][0], wpDr[r][0], 32, 32, 32, 32, 27);
    pack(dw[1 + r][3], wpDr[r][1], 32, 32, 32, 32, 27);
  }
  pack(cls_w1, wpCls1, 32, 32, 32, 32, 27);
  pack(cls_w2, wpCls2, 32, 32, 1, 16, 27);
  for (int i = 0; i < 7; ++i)
    pack(ctx_w[i], wpCtx[i], ctxCin[i], ctxCpi[i], ctxCout[i], ctxNp[i], 9);

  scale_init_k<<<(VOX + 255) / 256, 256, 0, stream>>>(costs, nullptr, 0.f, VOX);

  // ---- feature extraction (ref + 2 targets) ----
  const float* imgs[3] = {ref, targets, targets + (size_t)3 * Himg * Wimg};
  for (int i = 0; i < 3; ++i) {
    int tot = Himg * Wimg * 32;
    chlast_k<<<(tot + 255) / 256, 256, 0, stream>>>(imgs[i], bufImg,
                                                    Himg * Wimg, 3, 32);
    conv(bufImg, wpF[0], nullptr, nullptr, nullptr, pA, nullptr,
         1, 256, 320, 32, 1, 128, 160, 32, 32, 32, 1, 3, 3, 1, 2, 1, 1.f);
    conv(pA, wpF[1], nullptr, nullptr, nullptr, pB, nullptr,
         1, 128, 160, 32, 1, 128, 160, 32, 32, 32, 1, 3, 3, 1, 1, 1, 1.f);
    conv(pB, wpF[2], nullptr, nullptr, nullptr, pC, nullptr,
         1, 128, 160, 32, 1, 64, 80, 32, 32, 32, 1, 3, 3, 1, 2, 1, 1.f);
    conv(pC, wpF[3], nullptr, nullptr, nullptr, feats[i], nullptr,
         1, 64, 80, 32, 1, 64, 80, 32, 32, 32, 1, 3, 3, 1, 1, 0, 1.f);
  }

  // ---- per-target: cost volume + 3D regularization ----
  for (int j = 0; j < 2; ++j) {
    warp_cost_k<<<(VOX + 255) / 256, 256, 0, stream>>>(
        feats[0], feats[1 + j], pose, Kmat, Kinv, costvol, j, L, h, w);
    conv(costvol, wpD0a, dw[0][1], dw[0][2], nullptr, volA, nullptr,
         L, h, w, 64, L, h, w, 32, 32, 32, 3, 3, 3, 1, 1, 1, BN_INV_C);
    conv(volA, wpD0b, dw[0][4], dw[0][5], nullptr, volB, nullptr,
         L, h, w, 32, L, h, w, 32, 32, 32, 3, 3, 3, 1, 1, 1, BN_INV_C);
    for (int r = 0; r < 4; ++r) {
      conv(volB, wpDr[r][0], dw[1 + r][1], dw[1 + r][2], nullptr, volA, nullptr,
           L, h, w, 32, L, h, w, 32, 32, 32, 3, 3, 3, 1, 1, 1, BN_INV_C);
      conv(volA, wpDr[r][1], dw[1 + r][4], dw[1 + r][5], volB, volB, nullptr,
           L, h, w, 32, L, h, w, 32, 32, 32, 3, 3, 3, 1, 1, 0, BN_INV_C);
    }
    conv(volB, wpCls1, cls_g, cls_b, nullptr, volA, nullptr,
         L, h, w, 32, L, h, w, 32, 32, 32, 3, 3, 3, 1, 1, 1, BN_INV_C);
    conv(volA, wpCls2, nullptr, nullptr, nullptr, nullptr, costs,
         L, h, w, 32, L, h, w, 1, 16, 16, 3, 3, 3, 1, 1, 0, 1.f);
  }

  // ---- context network over depth slices (KD=1 conv, 16-slice chunks) ----
  {
    int tot = VOX * 64;
    ctx_input_k<<<(tot + 255) / 256, 256, 0, stream>>>(feats[0], costs,
                                                       costvol, L, HWs);
  }
  scale_init_k<<<(VOX + 255) / 256, 256, 0, stream>>>(refined, costs, 0.5f,
                                                      VOX);
  for (int c = 0; c < 4; ++c) {
    const _Float16* cin = costvol + (size_t)c * 16 * HWs * 64;
    float* rout = refined + (size_t)c * 16 * HWs;
    _Float16* pp[2] = {volA, volB};
    const _Float16* cur = cin;
    int CpiCur = 64;
    for (int i = 0; i < 6; ++i) {
      conv(cur, wpCtx[i], nullptr, nullptr, nullptr, pp[i & 1], nullptr,
           16, h, w, CpiCur, 16, h, w, ctxCout[i], ctxCout[i], ctxNp[i],
           1, 3, 3, ctxDil[i], 1, 2, 1.f);
      cur = pp[i & 1];
      CpiCur = ctxCout[i];
    }
    conv(cur, wpCtx[6], nullptr, nullptr, nullptr, nullptr, rout,
         16, h, w, CpiCur, 16, h, w, 1, 16, 16, 1, 3, 3, ctxDil[6], 1, 2, 1.f);
  }

  // ---- upsample + softmax + disparity -> depth ----
  final_k<<<(Himg * Wimg + 255) / 256, 256, 0, stream>>>(
      refined, (float*)d_out, L, h, w, Himg, Wimg);
}
